// ATGCO_16303695856134
// MI455X (gfx1250) — matrix operand order repaired and checked by reading, not compiled, not run on hardware
//
#include <hip/hip_runtime.h>
#include <stdint.h>

// ---------- types ----------
typedef __attribute__((ext_vector_type(16))) __bf16 v16bf;
typedef __attribute__((ext_vector_type(8)))  __bf16 v8bf;
typedef __attribute__((ext_vector_type(8)))  float  v8f;
typedef __attribute__((ext_vector_type(4)))  unsigned u32x4;
typedef __attribute__((ext_vector_type(8)))  unsigned u32x8;

union V16 { v16bf v; v8bf h[2]; __bf16 e[16]; };
union V8F { v8f  v; float e[8]; };

__device__ __forceinline__ void split_bf16(float x, __bf16& hi, __bf16& lo) {
  hi = (__bf16)x;
  lo = (__bf16)(x - (float)hi);
}

// ---- async DMA: global -> LDS, 16B, tracked by ASYNCcnt ----
__device__ __forceinline__ void async_ld_b128(void* dst_lds, const void* src) {
  unsigned lds = (unsigned)(uintptr_t)dst_lds;  // low 32 bits of flat shared addr == LDS offset
  asm volatile("global_load_async_to_lds_b128 %0, %1, off"
               :: "v"(lds), "v"(src) : "memory");
}
__device__ __forceinline__ void wait_async0() {
  asm volatile("s_wait_asynccnt 0" ::: "memory");
}

// ---- TDM: 2D tile DMA global -> LDS, tracked by TENSORcnt ----
// D# group0/group1 built per CDNA5 ISA ch.8; 2-group form (<=2D), issued once per wave.
__device__ __forceinline__ void tdm_load_2d(void* lds_dst, const void* gsrc,
                                            unsigned tile_w, unsigned tile_h,
                                            unsigned row_stride /* elems of 2B */) {
  unsigned lds = (unsigned)(uintptr_t)lds_dst;
  unsigned long long ga = (unsigned long long)(uintptr_t)gsrc;
  u32x4 g0;
  g0[0] = 1u;                                   // count=1, user descriptor
  g0[1] = lds;                                  // lds_addr (bytes)
  g0[2] = (unsigned)(ga & 0xffffffffu);         // global_addr[31:0]
  g0[3] = (unsigned)((ga >> 32) & 0x01ffffffu)  // global_addr[56:32]
        | (2u << 30);                           // type=2 ("image")
  u32x8 g1;
  g1[0] = (1u << 16);                           // data_size=1 -> 2-byte elems
  g1[1] = (row_stride & 0xffffu) << 16;         // tensor_dim0[15:0]  (bits 63:48)
  g1[2] = ((row_stride >> 16) & 0xffffu)        // tensor_dim0[31:16]
        | (0xffffu << 16);                      // tensor_dim1[15:0] = huge (no OOB clamp)
  g1[3] = (tile_w & 0xffffu) << 16;             // tensor_dim1[31:16]=0 | tile_dim0
  g1[4] = (tile_h & 0xffffu);                   // tile_dim1 | tile_dim2=0
  g1[5] = row_stride;                           // tensor_dim0_stride[31:0]
  g1[6] = 0u;                                   // stride[47:32]=0 | dim1_stride[15:0]=0
  g1[7] = 0u;                                   // dim1_stride[47:16]=0
  asm volatile("tensor_load_to_lds %0, %1" :: "s"(g0), "s"(g1) : "memory");
}

// ---------- sizes ----------
#define NB   8
#define NN   512
#define NF   256
#define NO   256
#define NK   16
// main GEMM per batch: M=512(i) x N=4096(o,k) x K=8192(j,s)

// ============================================================
// Kernel 0: split adj (f32) into bf16 hi/lo planes in workspace
// ============================================================
__global__ __launch_bounds__(256) void k_split_adj(const float* __restrict__ adj,
                                                   __bf16* __restrict__ ah,
                                                   __bf16* __restrict__ al,
                                                   long n4) {
  long t = (long)blockIdx.x * 256 + threadIdx.x;
  if (t >= n4) return;
  float4 v = ((const float4*)adj)[t];
  union { __bf16 e[4]; uint2 u; } H, L;
  split_bf16(v.x, H.e[0], L.e[0]);
  split_bf16(v.y, H.e[1], L.e[1]);
  split_bf16(v.z, H.e[2], L.e[2]);
  split_bf16(v.w, H.e[3], L.e[3]);
  ((uint2*)ah)[t] = H.u;
  ((uint2*)al)[t] = L.u;
}

// ============================================================
// Kernel 1: y[b,j,o,k] = sum_f x[b,j,f,k] * W[f,o]  (bf16x3 WMMA)
// x block staged through the async-to-LDS path.
// ============================================================
__global__ __launch_bounds__(256) void k_xw(const float* __restrict__ x,
                                            const float* __restrict__ w,
                                            __bf16* __restrict__ yh,
                                            __bf16* __restrict__ yl) {
  __shared__ __align__(16) float xs[32 * 16];   // one K-chunk of x: 32 f rows x 16 k

  const int b = blockIdx.y, j = blockIdx.x;
  const int t = threadIdx.x, lane = t & 31, wv = t >> 5;
  const int m = lane & 15, half = lane >> 4;

  const size_t xbase = ((size_t)(b * NN + j)) * NF * NK;

  V8F acc[2];
#pragma unroll
  for (int ot = 0; ot < 2; ++ot)
#pragma unroll
    for (int r = 0; r < 8; ++r) acc[ot].e[r] = 0.0f;

  for (int f0 = 0; f0 < NF; f0 += 32) {
    __syncthreads();
    if (t < 128)   // 128 x 16B async copies fill the 2 KB block
      async_ld_b128(&xs[t * 4], x + xbase + (size_t)f0 * NK + t * 4);
    wait_async0();
    __syncthreads();

    // B fragment: B[K=half*16+q, n=k] = x[f0+K, k]
    V16 Bh, Bl;
#pragma unroll
    for (int q = 0; q < 16; ++q) {
      int kk = half * 16 + q;
      split_bf16(xs[kk * 16 + m], Bh.e[q], Bl.e[q]);
    }

#pragma unroll
    for (int ot = 0; ot < 2; ++ot) {
      const int o0 = (wv * 2 + ot) * 16;
      // A fragment: A[m, K] = W[f0+K, o0+m]
      V16 Ah, Al;
#pragma unroll
      for (int q = 0; q < 16; ++q) {
        int kk = half * 16 + q;
        split_bf16(w[(size_t)(f0 + kk) * NO + (o0 + m)], Ah.e[q], Al.e[q]);
      }
      acc[ot].v = __builtin_amdgcn_wmma_f32_16x16x32_bf16(false, Ah.v, false, Bh.v, (short)0, acc[ot].v, false, false);
      acc[ot].v = __builtin_amdgcn_wmma_f32_16x16x32_bf16(false, Ah.v, false, Bl.v, (short)0, acc[ot].v, false, false);
      acc[ot].v = __builtin_amdgcn_wmma_f32_16x16x32_bf16(false, Al.v, false, Bh.v, (short)0, acc[ot].v, false, false);
    }
  }

#pragma unroll
  for (int ot = 0; ot < 2; ++ot) {
    const int o0 = (wv * 2 + ot) * 16;
#pragma unroll
    for (int r = 0; r < 8; ++r) {
      int o_abs = o0 + half * 8 + r;
      size_t off = (((size_t)b * NN + j) * NO + o_abs) * NK + m;
      __bf16 hi, lo;
      split_bf16(acc[ot].e[r], hi, lo);
      yh[off] = hi;
      yl[off] = lo;
    }
  }
}

// ============================================================
// Kernel 2: out[b,i,o,k] = sum_{j,s} adj[b,i,j,s] * y[b,j,o,(k-s)%16]
// Double-buffered: TDM streams adj tiles (128 x 32 bf16) into LDS while
// waves WMMA out of the other buffer; circulant y data is pre-swizzled
// into WMMA-fragment layout so the hot loop is pure ds_load_b128 + wmma.
// ============================================================
#define NJC (NN / 2)   // 256 K-chunks of 32 (2 j values each)

__global__ __launch_bounds__(256) void k_tprod(const __bf16* __restrict__ ah,
                                               const __bf16* __restrict__ al,
                                               const __bf16* __restrict__ yh,
                                               const __bf16* __restrict__ yl,
                                               float* __restrict__ out) {
  __shared__ __align__(16) __bf16 Ah_s[2][128][32];     // 16 KB (both buffers)
  __shared__ __align__(16) __bf16 Al_s[2][128][32];     // 16 KB
  __shared__ __align__(16) __bf16 Bfh[2][4][32][16];    //  8 KB  per-lane fragment layout
  __shared__ __align__(16) __bf16 Bfl[2][4][32][16];    //  8 KB

  const int b = blockIdx.z, i0 = blockIdx.y * 128, oa0 = blockIdx.x * 4;
  const int t = threadIdx.x, lane = t & 31, wv = t >> 5;
  const int m = lane & 15, half = lane >> 4;
  const int wi = wv & 3, wo = wv >> 2;   // 4 i-groups x 2 o-groups of waves

  V8F acc[2][2];
#pragma unroll
  for (int is = 0; is < 2; ++is)
#pragma unroll
    for (int oo = 0; oo < 2; ++oo)
#pragma unroll
      for (int r = 0; r < 8; ++r) acc[is][oo].e[r] = 0.0f;

  const size_t arow = (size_t)NN * NK;  // 8192 elems per adj row
  const __bf16* abase_h = ah + (size_t)(b * NN + i0) * arow;
  const __bf16* abase_l = al + (size_t)(b * NN + i0) * arow;

  // stage buffer nb with K-chunk starting at j0 (2 j values)
  auto stage = [&](int nb, int j0) {
    if (wv == 0) {   // one wave drives the Tensor Data Mover
      tdm_load_2d(&Ah_s[nb][0][0], abase_h + (size_t)j0 * NK, 32u, 128u, (unsigned)arow);
      tdm_load_2d(&Al_s[nb][0][0], abase_l + (size_t)j0 * NK, 32u, 128u, (unsigned)arow);
    }
    // pre-swizzle circulant y into fragment layout:
    // Bfrag[o][lane][q] = y[b, j0+(lane>>4), oa0+o, ((lane&15)-q) & 15]
    const int arr  = t >> 7;          // 0: hi plane, 1: lo plane
    const int slot = t & 127;
    const int o = slot >> 5, ln = slot & 31;
    const int jl = ln >> 4, n = ln & 15;
    const __bf16* ysrc = (arr ? yl : yh)
        + (((size_t)b * NN + (j0 + jl)) * NO + (oa0 + o)) * NK;
    V16 f;
#pragma unroll
    for (int s = 0; s < 16; ++s) f.e[s] = ysrc[(n - s) & 15];
    __bf16* dst = arr ? &Bfl[nb][o][ln][0] : &Bfh[nb][o][ln][0];
    *(v8bf*)(dst)     = f.h[0];
    *(v8bf*)(dst + 8) = f.h[1];
  };

  stage(0, 0);
  int p = 0;
  for (int jc = 0; jc < NJC; ++jc) {
    __builtin_amdgcn_s_wait_tensorcnt(0);   // wave0's TDM for buf p complete
    __syncthreads();                        // publish buf p to all waves
    if (jc + 1 < NJC) stage(p ^ 1, (jc + 1) * 2);
    if (jc + 2 < NJC)                       // warm next y slice in GL2
      __builtin_prefetch(yh + (((size_t)b * NN + (jc + 2) * 2) * NO + oa0) * NK, 0, 1);

    // ---- compute from buffer p: 16 ds_load_b128 feed 12 WMMAs per wave ----
    V16 B_h[2], B_l[2];
#pragma unroll
    for (int oo = 0; oo < 2; ++oo) {
      const int o = wo * 2 + oo;
      B_h[oo].h[0] = *(const v8bf*)&Bfh[p][o][lane][0];
      B_h[oo].h[1] = *(const v8bf*)&Bfh[p][o][lane][8];
      B_l[oo].h[0] = *(const v8bf*)&Bfl[p][o][lane][0];
      B_l[oo].h[1] = *(const v8bf*)&Bfl[p][o][lane][8];
    }
#pragma unroll
    for (int is = 0; is < 2; ++is) {
      const int row = wi * 32 + is * 16 + m;
      V16 A_h, A_l;
      A_h.h[0] = *(const v8bf*)&Ah_s[p][row][half * 16];
      A_h.h[1] = *(const v8bf*)&Ah_s[p][row][half * 16 + 8];
      A_l.h[0] = *(const v8bf*)&Al_s[p][row][half * 16];
      A_l.h[1] = *(const v8bf*)&Al_s[p][row][half * 16 + 8];
#pragma unroll
      for (int oo = 0; oo < 2; ++oo) {
        acc[is][oo].v = __builtin_amdgcn_wmma_f32_16x16x32_bf16(false, A_h.v, false, B_h[oo].v, (short)0, acc[is][oo].v, false, false);
        acc[is][oo].v = __builtin_amdgcn_wmma_f32_16x16x32_bf16(false, A_h.v, false, B_l[oo].v, (short)0, acc[is][oo].v, false, false);
        acc[is][oo].v = __builtin_amdgcn_wmma_f32_16x16x32_bf16(false, A_l.v, false, B_h[oo].v, (short)0, acc[is][oo].v, false, false);
      }
    }
    p ^= 1;
  }

  // ---- store: C element (r,lane): i = half*8+r within tile, k = lane&15 ----
#pragma unroll
  for (int is = 0; is < 2; ++is)
#pragma unroll
    for (int oo = 0; oo < 2; ++oo)
#pragma unroll
      for (int r = 0; r < 8; ++r) {
        int i_out = i0 + wi * 32 + is * 16 + half * 8 + r;
        int o_abs = oa0 + wo * 2 + oo;
        size_t off = (((size_t)b * NN + i_out) * NO + o_abs) * NK + m;
        out[off] = acc[is][oo].e[r];
      }
}

// ============================================================
extern "C" void kernel_launch(void* const* d_in, const int* in_sizes, int n_in,
                              void* d_out, int out_size, void* d_ws, size_t ws_size,
                              hipStream_t stream) {
  const float* x   = (const float*)d_in[0];   // [8,512,256,16]
  const float* adj = (const float*)d_in[1];   // [8,512,512,16]
  const float* w   = (const float*)d_in[2];   // [256,256]
  float* out = (float*)d_out;                 // [8,512,256,16]

  char* ws = (char*)d_ws;
  const size_t adjN = (size_t)NB * NN * NN * NK;   // 33,554,432
  const size_t yN   = (size_t)NB * NN * NO * NK;   // 16,777,216
  __bf16* ah = (__bf16*)(ws);
  __bf16* al = (__bf16*)(ws + adjN * 2);
  __bf16* yh = (__bf16*)(ws + adjN * 4);
  __bf16* yl = (__bf16*)(ws + adjN * 4 + yN * 2);

  long n4 = (long)(adjN / 4);
  k_split_adj<<<dim3((unsigned)((n4 + 255) / 256)), 256, 0, stream>>>(adj, ah, al, n4);
  k_xw<<<dim3(NN, NB), 256, 0, stream>>>(x, w, yh, yl);
  k_tprod<<<dim3(NO / 4, NN / 128, NB), 256, 0, stream>>>(ah, al, yh, yl, out);
}
